// RFNTModel_68968584839493
// MI455X (gfx1250) — matrix operand order repaired and checked by reading, not compiled, not run on hardware
//
#include <hip/hip_runtime.h>
#include <stdint.h>

typedef __bf16 bf16;
typedef __attribute__((ext_vector_type(16))) __bf16 v16bf;
typedef __attribute__((ext_vector_type(8)))  float  v8f;

#define BATCH 32
#define EPSV  1e-5f

union BFrag { v16bf v; uint32_t u[8]; };

// ---------------------------------------------------------------- elementwise
__global__ void k_add(const float* __restrict__ a, const float* __restrict__ b,
                      float* __restrict__ o, int n) {
  int i = blockIdx.x * blockDim.x + threadIdx.x;
  if (i < n) o[i] = a[i] + b[i];
}

// ------------------------------------------------------- weight prep (f32->bf16, K-major)
// A[m][k], k=(cin*KH*KW + ky*KW + kx), w is OIHW (Cout,Cin,KH,KW); rows padded to Mpad
__global__ void k_prep_w(const float* __restrict__ w, bf16* __restrict__ A,
                         int Cout, int Cin, int KH, int KW, int Kpad, int Mpad) {
  int t = blockIdx.x * blockDim.x + threadIdx.x;
  if (t >= Mpad * Kpad) return;
  int m = t / Kpad, k = t % Kpad;
  int Keff = Cin * KH * KW;
  float v = 0.f;
  if (m < Cout && k < Keff) {
    int cin = k / (KH * KW), r = k % (KH * KW);
    int ky = r / KW, kx = r % KW;
    v = w[((m * Cin + cin) * KH + ky) * KW + kx];
  }
  A[t] = (bf16)v;
}

// deconv parity-class weights: A[m][cin*4 + kyi*2 + kxi] = w[cin][m][3-(py+2kyi)][3-(px+2kxi)]
// (w is dec_wt: (128,64,4,4); reference uses wt = flip(transpose(w)))
__global__ void k_prep_wdec(const float* __restrict__ w, bf16* __restrict__ A,
                            int py, int px) {
  int t = blockIdx.x * blockDim.x + threadIdx.x;
  if (t >= 64 * 512) return;
  int m = t / 512, k = t % 512;
  int cin = k >> 2, kyi = (k >> 1) & 1, kxi = k & 1;
  int ky = py + 2 * kyi, kx = px + 2 * kxi;
  A[t] = (bf16)w[((cin * 64 + m) * 4 + (3 - ky)) * 4 + (3 - kx)];
}

// weT[d][k] = embed_w[k][d], k padded 588->608
__global__ void k_prep_embT(const float* __restrict__ we, bf16* __restrict__ weT) {
  int t = blockIdx.x * blockDim.x + threadIdx.x;
  if (t >= 768 * 608) return;
  int d = t / 608, k = t % 608;
  weT[t] = (bf16)((k < 588) ? we[k * 768 + d] : 0.f);
}

// ---------------------------------------------------------------- im2col (per image)
template <typename T>
__global__ void k_im2col(const T* __restrict__ in, bf16* __restrict__ col,
                         int Cin, int Hin, int Win, int Hout, int Wout,
                         int KH, int KW, int stride, int pad, int Kpad) {
  long long t = (long long)blockIdx.x * blockDim.x + threadIdx.x;
  long long total = (long long)Hout * Wout * Kpad;
  if (t >= total) return;
  int k = (int)(t % Kpad);
  long long n = t / Kpad;
  int Keff = Cin * KH * KW;
  float v = 0.f;
  if (k < Keff) {
    int cin = k / (KH * KW), r = k % (KH * KW);
    int ky = r / KW, kx = r % KW;
    int oy = (int)(n / Wout), ox = (int)(n % Wout);
    int iy = oy * stride - pad + ky, ix = ox * stride - pad + kx;
    if (iy >= 0 && iy < Hin && ix >= 0 && ix < Win)
      v = (float)in[((long long)cin * Hin + iy) * Win + ix];
  }
  col[t] = (bf16)v;
}

// deconv parity-class im2col: one 224x224 output parity class (py,px) over a
// (128,112,112) image; col[(y*112+x)*512 + cin*4 + kyi*2 + kxi]
__global__ void k_im2col_dec(const bf16* __restrict__ in, bf16* __restrict__ col,
                             int py, int px) {
  long long t = (long long)blockIdx.x * blockDim.x + threadIdx.x;
  if (t >= (long long)12544 * 512) return;
  int k = (int)(t & 511);
  int n = (int)(t >> 9);
  int cin = k >> 2, kyi = (k >> 1) & 1, kxi = k & 1;
  int y = n / 112, x = n % 112;
  int iy = y + py - 1 + kyi;
  int ix = x + px - 1 + kxi;
  float v = 0.f;
  if (iy >= 0 && iy < 112 && ix >= 0 && ix < 112)
    v = (float)in[(cin * 112 + iy) * 112 + ix];
  col[t] = (bf16)v;
}

// ---------------------------------------------------------------- WMMA GEMM (64x16 per wave)
// D[M][N] = A[M][Kpad] * B[N][Kpad]^T + bias. A padded to 64-row multiples.
// Each wave: 4 M-tiles x 1 N-tile. All 5 fragments (1 B + 4 A) are loaded into
// independent registers before the WMMA block so the scheduler can overlap
// loads for tile i+1..3 with the WMMA on tile i (partial s_wait_loadcnt).
// scatter=1: deconv epilogue, out[m][(2y+py)*224 + 2x+px] with n=(y*112+x), ld=224*224.
__global__ void k_gemm4(const bf16* __restrict__ A, const bf16* __restrict__ Bm,
                        const float* __restrict__ bias, void* __restrict__ outp,
                        int Mact, int N, int Kpad, int act, int out_f32, int bias_per_n,
                        int scatter, int py, int px) {
  int wave = threadIdx.x >> 5;
  int lane = threadIdx.x & 31;
  int tilesN = N >> 4;
  int ntile = blockIdx.x * 8 + wave;
  if (ntile >= tilesN) return;              // wave-uniform exit, EXEC stays full
  int m0 = blockIdx.y << 6, n0 = ntile << 4;
  int half = lane >> 4, l15 = lane & 15;

  const uint32_t* brow  = (const uint32_t*)(Bm + (size_t)(n0 + l15) * Kpad);
  const uint32_t* arow0 = (const uint32_t*)(A  + (size_t)(m0 + l15) * Kpad);
  const size_t astr = (size_t)16 * (Kpad >> 1);   // uints per 16 A-rows
  int ha = half * 4, hb = half * 8;

  v8f z = {};
  v8f acc0 = z, acc1 = z, acc2 = z, acc3 = z;
  for (int kb = 0; kb < Kpad; kb += 32) {
    const uint32_t* bp  = brow  + (kb >> 1);
    const uint32_t* ap0 = arow0 + (kb >> 1);
    const uint32_t* ap1 = ap0 + astr;
    const uint32_t* ap2 = ap1 + astr;
    const uint32_t* ap3 = ap2 + astr;
    BFrag fb, fa0, fa1, fa2, fa3;
#pragma unroll
    for (int j = 0; j < 8; ++j) fb.u[j] = bp[(j < 4 ? ha : 8 + ha - 4) + j];
#pragma unroll
    for (int v = 0; v < 4; ++v) {
      fa0.u[v] = ap0[ha + v]; fa0.u[4 + v] = ap0[8 + ha + v];
      fa1.u[v] = ap1[ha + v]; fa1.u[4 + v] = ap1[8 + ha + v];
      fa2.u[v] = ap2[ha + v]; fa2.u[4 + v] = ap2[8 + ha + v];
      fa3.u[v] = ap3[ha + v]; fa3.u[4 + v] = ap3[8 + ha + v];
    }
    __builtin_prefetch(bp + 16, 0, 1);            // next k-step of this B row
    acc0 = __builtin_amdgcn_wmma_f32_16x16x32_bf16(false, fa0.v, false, fb.v,
                                                   (short)0, acc0, false, false);
    acc1 = __builtin_amdgcn_wmma_f32_16x16x32_bf16(false, fa1.v, false, fb.v,
                                                   (short)0, acc1, false, false);
    acc2 = __builtin_amdgcn_wmma_f32_16x16x32_bf16(false, fa2.v, false, fb.v,
                                                   (short)0, acc2, false, false);
    acc3 = __builtin_amdgcn_wmma_f32_16x16x32_bf16(false, fa3.v, false, fb.v,
                                                   (short)0, acc3, false, false);
  }

  int n = n0 + l15;
  size_t ldOut = N, outcol = n;
  if (scatter) {
    int y = n / 112, xq = n % 112;
    outcol = (size_t)(2 * y + py) * 224 + 2 * xq + px;
    ldOut = (size_t)224 * 224;
  }
#pragma unroll
  for (int mt = 0; mt < 4; ++mt) {
    v8f a = (mt == 0) ? acc0 : (mt == 1) ? acc1 : (mt == 2) ? acc2 : acc3;
#pragma unroll
    for (int r = 0; r < 8; ++r) {
      int m = m0 + mt * 16 + half * 8 + r;
      if (m >= Mact) continue;
      float v = a[r] + (bias_per_n ? bias[n] : bias[m]);
      if (act == 1) v = fmaxf(v, 0.f);
      else if (act == 2) v = tanhf(v);
      size_t o = (size_t)m * ldOut + outcol;
      if (out_f32) ((float*)outp)[o] = v;
      else         ((bf16*)outp)[o] = (bf16)v;
    }
  }
}

// ---------------------------------------------------------------- batchnorm over (B,H,W)
__global__ void k_bn_stats(const bf16* __restrict__ h, float* __restrict__ mu,
                           float* __restrict__ rstd, int C, int HW) {
  __shared__ float ss[256], sq[256];
  int c = blockIdx.x, tid = threadIdx.x;
  long long n = (long long)BATCH * HW;
  float s = 0.f, q = 0.f;
  for (long long j = tid; j < n; j += 256) {
    int b = (int)(j / HW), i = (int)(j % HW);
    float v = (float)h[((long long)b * C + c) * HW + i];
    s += v; q += v * v;
  }
  ss[tid] = s; sq[tid] = q; __syncthreads();
  for (int st = 128; st > 0; st >>= 1) {
    if (tid < st) { ss[tid] += ss[tid + st]; sq[tid] += sq[tid + st]; }
    __syncthreads();
  }
  if (tid == 0) {
    float m = ss[0] / (float)n;
    float var = sq[0] / (float)n - m * m;
    mu[c] = m; rstd[c] = rsqrtf(var + EPSV);
  }
}

__global__ void k_bn_relu(const bf16* __restrict__ h, bf16* __restrict__ o,
                          const float* __restrict__ mu, const float* __restrict__ rstd,
                          const float* __restrict__ g, const float* __restrict__ bb,
                          int C, int HW, long long total) {
  long long t = (long long)blockIdx.x * blockDim.x + threadIdx.x;
  if (t >= total) return;
  int c = (int)((t / HW) % C);
  float v = ((float)h[t] - mu[c]) * rstd[c] * g[c] + bb[c];
  o[t] = (bf16)fmaxf(v, 0.f);
}

// ---------------------------------------------------------------- mean patch (64 x 608 bf16, rows>=32 zero)
__global__ void k_mean_patch(const float* __restrict__ x, bf16* __restrict__ mp) {
  int t = blockIdx.x * blockDim.x + threadIdx.x;
  if (t >= 64 * 608) return;
  int b = t / 608, k = t % 608;
  float v = 0.f;
  if (b < BATCH && k < 588) {
    int c = k / 196, r = k % 196, py = r / 14, px = r % 14;
    const float* base = x + ((long long)b * 3 + c) * 224 * 224;
    float s = 0.f;
    for (int gy = 0; gy < 16; ++gy)
      for (int gx = 0; gx < 16; ++gx)
        s += base[(gy * 14 + py) * 224 + gx * 14 + px];
    v = s * (1.f / 256.f);
  }
  mp[t] = (bf16)v;
}

// ---------------------------------------------------------------- cosine-sim rows
__global__ void k_rowcs(const float* __restrict__ f1, const float* __restrict__ f2,
                        float* __restrict__ cs) {
  __shared__ float a1[256], a2[256], b1[256], b2[256];
  int b = blockIdx.x, tid = threadIdx.x;
  const float* r1 = f1 + b * 768;
  const float* r2 = f2 + b * 768;
  float s1 = 0, s2 = 0, q1 = 0, q2 = 0;
  for (int d = tid; d < 768; d += 256) {
    float v1 = r1[d], v2 = r2[d];
    s1 += v1; q1 += v1 * v1; s2 += v2; q2 += v2 * v2;
  }
  a1[tid] = s1; b1[tid] = q1; a2[tid] = s2; b2[tid] = q2; __syncthreads();
  for (int st = 128; st > 0; st >>= 1) {
    if (tid < st) { a1[tid] += a1[tid+st]; b1[tid] += b1[tid+st];
                    a2[tid] += a2[tid+st]; b2[tid] += b2[tid+st]; }
    __syncthreads();
  }
  __shared__ float m1s, m2s, inv;
  if (tid == 0) {
    const float n = 768.f;
    float m1 = a1[0] / n, m2 = a2[0] / n;
    float sd1 = sqrtf((b1[0] - n * m1 * m1) / (n - 1.f));
    float sd2 = sqrtf((b2[0] - n * m2 * m2) / (n - 1.f));
    m1s = m1; m2s = m2; inv = 1.f / ((n - 1.f) * sd1 * sd2);
  }
  __syncthreads();
  for (int d = tid; d < 768; d += 256)
    cs[b * 768 + d] = (r1[d] - m1s) * (r2[d] - m2s) * inv;
}

// ---------------------------------------------------------------- batchnorm over axis 0 + write cs
__global__ void k_bnorm_cols(const float* __restrict__ cs, const float* __restrict__ g,
                             const float* __restrict__ bb, float* __restrict__ out) {
  int d = blockIdx.x * blockDim.x + threadIdx.x;
  if (d >= 768) return;
  float s = 0, q = 0;
  for (int b = 0; b < BATCH; ++b) { float v = cs[b * 768 + d]; s += v; q += v * v; }
  float m = s / (float)BATCH;
  float var = q / (float)BATCH - m * m;
  float r = rsqrtf(var + EPSV);
  for (int b = 0; b < BATCH; ++b)
    out[b * 768 + d] = (cs[b * 768 + d] - m) * r * g[d] + bb[d];
}

// ---------------------------------------------------------------- classifier head
__global__ void k_pred(const float* __restrict__ cs, const float* __restrict__ w,
                       const float* __restrict__ bias, float* __restrict__ pred) {
  __shared__ float sh[256];
  int b = blockIdx.x, tid = threadIdx.x;
  float s = 0.f;
  for (int d = tid; d < 768; d += 256) s += cs[b * 768 + d] * w[d];
  sh[tid] = s; __syncthreads();
  for (int st = 128; st > 0; st >>= 1) {
    if (tid < st) sh[tid] += sh[tid + st];
    __syncthreads();
  }
  if (tid == 0) pred[b] = sh[0] + bias[0];
}

// ================================================================ host
static inline size_t alup(size_t x) { return (x + 255) & ~(size_t)255; }

extern "C" void kernel_launch(void* const* d_in, const int* in_sizes, int n_in,
                              void* d_out, int out_size, void* d_ws, size_t ws_size,
                              hipStream_t stream) {
  (void)in_sizes; (void)n_in; (void)out_size; (void)ws_size;
  const float* x      = (const float*)d_in[0];
  const float* noise  = (const float*)d_in[1];
  const float* enc_w1 = (const float*)d_in[2];
  const float* enc_b1 = (const float*)d_in[3];
  const float* enc_w2 = (const float*)d_in[4];
  const float* enc_b2 = (const float*)d_in[5];
  const float* bn_g   = (const float*)d_in[6];
  const float* bn_b   = (const float*)d_in[7];
  const float* mid_w1 = (const float*)d_in[8];
  const float* mid_b1 = (const float*)d_in[9];
  const float* mid_w2 = (const float*)d_in[10];
  const float* mid_b2 = (const float*)d_in[11];
  const float* dec_wt = (const float*)d_in[12];
  const float* dec_bt = (const float*)d_in[13];
  const float* dec_w2 = (const float*)d_in[14];
  const float* dec_b2 = (const float*)d_in[15];
  const float* emb_w  = (const float*)d_in[16];
  const float* emb_b  = (const float*)d_in[17];
  const float* bng768 = (const float*)d_in[18];
  const float* bnb768 = (const float*)d_in[19];
  const float* cls_w  = (const float*)d_in[20];
  const float* cls_b  = (const float*)d_in[21];
  float* out = (float*)d_out;

  // ---- workspace carve-up (bytes); total ~510 MB
  char* p = (char*)d_ws;
  size_t off = 0;
  auto take = [&](size_t bytes) { char* r = p + off; off = alup(off + bytes); return (void*)r; };
  const size_t IMG  = (size_t)BATCH * 3 * 224 * 224;
  float* xn   = (float*)take(IMG * 4);
  float* xns  = (float*)take(IMG * 4);
  bf16*  h1   = (bf16*) take((size_t)BATCH * 64 * 224*224 * 2);   // also h5
  bf16*  h2   = (bf16*) take((size_t)BATCH * 128 * 112*112 * 2);  // also h3
  bf16*  h2n  = (bf16*) take((size_t)BATCH * 128 * 112*112 * 2);  // also h4
  bf16*  col  = (bf16*) take((size_t)50176 * 576 * 2);            // max im2col (conv1/final)
  bf16*  wA1  = (bf16*) take(64  * 32   * 2);
  bf16*  wA2  = (bf16*) take(128 * 576  * 2);
  bf16*  wAm1 = (bf16*) take(128 * 1152 * 2);
  bf16*  wAm2 = (bf16*) take(128 * 1152 * 2);
  bf16*  wAdp = (bf16*) take((size_t)4 * 64 * 512 * 2);           // deconv parity classes
  bf16*  wAf  = (bf16*) take(64  * 576  * 2);
  bf16*  weT  = (bf16*) take(768 * 608  * 2);
  bf16*  mp_o = (bf16*) take(64 * 608 * 2);
  bf16*  mp_n = (bf16*) take(64 * 608 * 2);
  float* f_o  = (float*)take(BATCH * 768 * 4);
  float* f_n  = (float*)take(BATCH * 768 * 4);
  float* csr  = (float*)take(BATCH * 768 * 4);
  float* mu   = (float*)take(128 * 4);
  float* rstd = (float*)take(128 * 4);
  bf16*  h5   = h1;
  bf16*  h3   = h2;
  bf16*  h4   = h2n;

  // ---- weight prep
  auto gsz = [](long long n) { return dim3((unsigned)((n + 255) / 256)); };
  k_prep_w<<<gsz(64*32),    256, 0, stream>>>(enc_w1, wA1, 64,  3,   3, 3, 32,   64);
  k_prep_w<<<gsz(128*576),  256, 0, stream>>>(enc_w2, wA2, 128, 64,  3, 3, 576,  128);
  k_prep_w<<<gsz(128*1152), 256, 0, stream>>>(mid_w1, wAm1,128, 128, 3, 3, 1152, 128);
  k_prep_w<<<gsz(128*1152), 256, 0, stream>>>(mid_w2, wAm2,128, 128, 3, 3, 1152, 128);
  k_prep_w<<<gsz(64*576),   256, 0, stream>>>(dec_w2, wAf, 3,   64,  3, 3, 576,  64);
  for (int cls = 0; cls < 4; ++cls)
    k_prep_wdec<<<gsz(64*512), 256, 0, stream>>>(dec_wt, wAdp + (size_t)cls * 64 * 512,
                                                 cls >> 1, cls & 1);
  k_prep_embT<<<gsz(768*608), 256, 0, stream>>>(emb_w, weT);

  // ---- x + noise
  k_add<<<gsz((long long)IMG), 256, 0, stream>>>(x, noise, xn, (int)IMG);

  // ---- generic conv layer: per-image im2col + WMMA GEMM (64x16 wave tiles)
  auto conv = [&](const void* in, int in_f32, int Cin, int Hin, int Win,
                  int KH, int KW, int stride, int pad,
                  const bf16* Aw, const float* bias, int Cout, int Mpad64, int Kpad,
                  void* o, int out_f32, int act, int Hout, int Wout) {
    long long Npix = (long long)Hout * Wout;
    long long colN = Npix * Kpad;
    dim3 gg((unsigned)(((Npix / 16) + 7) / 8), (unsigned)(Mpad64 / 64));
    for (int b = 0; b < BATCH; ++b) {
      const char* inb = (const char*)in + (size_t)b * Cin * Hin * Win * (in_f32 ? 4 : 2);
      if (in_f32)
        k_im2col<float><<<gsz(colN), 256, 0, stream>>>((const float*)inb, col,
            Cin, Hin, Win, Hout, Wout, KH, KW, stride, pad, Kpad);
      else
        k_im2col<bf16><<<gsz(colN), 256, 0, stream>>>((const bf16*)inb, col,
            Cin, Hin, Win, Hout, Wout, KH, KW, stride, pad, Kpad);
      char* ob = (char*)o + (size_t)b * Cout * Npix * (out_f32 ? 4 : 2);
      k_gemm4<<<gg, 256, 0, stream>>>(Aw, col, bias, ob, Cout, (int)Npix, Kpad,
                                      act, out_f32, 0, 0, 0, 0);
    }
  };

  // noise generator
  conv(xn,  1, 3,   224, 224, 3, 3, 1, 1, wA1,  enc_b1, 64,  64,  32,   h1,  0, 1, 224, 224);
  conv(h1,  0, 64,  224, 224, 3, 3, 2, 1, wA2,  enc_b2, 128, 128, 576,  h2,  0, 0, 112, 112);
  k_bn_stats<<<128, 256, 0, stream>>>(h2, mu, rstd, 128, 112 * 112);
  {
    long long tot = (long long)BATCH * 128 * 112 * 112;
    k_bn_relu<<<gsz(tot), 256, 0, stream>>>(h2, h2n, mu, rstd, bn_g, bn_b,
                                            128, 112 * 112, tot);
  }
  conv(h2n, 0, 128, 112, 112, 3, 3, 1, 1, wAm1, mid_b1, 128, 128, 1152, h3,  0, 1, 112, 112);
  conv(h3,  0, 128, 112, 112, 3, 3, 1, 1, wAm2, mid_b2, 128, 128, 1152, h4,  0, 1, 112, 112);

  // transposed conv as 4 parity-class GEMMs (K=512 each, no zero taps)
  {
    long long colN = (long long)12544 * 512;
    dim3 gg((unsigned)((784 + 7) / 8), 1);
    for (int b = 0; b < BATCH; ++b) {
      const bf16* inb = h4 + (size_t)b * 128 * 12544;
      bf16* ob = h5 + (size_t)b * 64 * 50176;
      for (int cls = 0; cls < 4; ++cls) {
        int py = cls >> 1, px = cls & 1;
        k_im2col_dec<<<gsz(colN), 256, 0, stream>>>(inb, col, py, px);
        k_gemm4<<<gg, 256, 0, stream>>>(wAdp + (size_t)cls * 64 * 512, col, dec_bt, ob,
                                        64, 12544, 512, 1, 0, 0, 1, py, px);
      }
    }
  }

  conv(h5,  0, 64,  224, 224, 3, 3, 1, 1, wAf,  dec_b2, 3,   64,  576,  xns, 1, 2, 224, 224);

  // patch embedding via mean-patch (token-mean is linear)
  k_mean_patch<<<gsz(64 * 608), 256, 0, stream>>>(x,   mp_o);
  k_mean_patch<<<gsz(64 * 608), 256, 0, stream>>>(xns, mp_n);
  {
    dim3 gg((unsigned)(((768 / 16) + 7) / 8), 1);
    k_gemm4<<<gg, 256, 0, stream>>>(mp_o, weT, emb_b, f_o, BATCH, 768, 608, 0, 1, 1, 0, 0, 0);
    k_gemm4<<<gg, 256, 0, stream>>>(mp_n, weT, emb_b, f_n, BATCH, 768, 608, 0, 1, 1, 0, 0, 0);
  }

  // head
  k_rowcs<<<BATCH, 256, 0, stream>>>(f_o, f_n, csr);
  k_bnorm_cols<<<3, 256, 0, stream>>>(csr, bng768, bnb768, out + BATCH);
  k_pred<<<BATCH, 256, 0, stream>>>(out + BATCH, cls_w, cls_b, out);
}
